// GCN_CSA_Block_62130996904363
// MI455X (gfx1250) — matrix operand order repaired and checked by reading, not compiled, not run on hardware
//
#include <hip/hip_runtime.h>
#include <math.h>

typedef __attribute__((ext_vector_type(16))) _Float16 v16h;
typedef __attribute__((ext_vector_type(8)))  _Float16 v8h;
typedef __attribute__((ext_vector_type(8)))  float    v8f;

#define B_  8
#define C_  64
#define L_  2048
#define U_  10

// ---------------------------------------------------------------------------
// Kernel 1: transpose-read x, build qn (f16, L2-normalized rows), and the
// first GEMM's B-panel p1t = (qf@W1)^T stored K-major (B,16,L) f16 so the
// WMMA B operand is a contiguous 32B load. Also init deg=1 (the +I term).
// ---------------------------------------------------------------------------
__global__ void k_prep(const float* __restrict__ x, const float* __restrict__ W1,
                       _Float16* __restrict__ qn, _Float16* __restrict__ p1t,
                       unsigned* __restrict__ deg) {
  int t = blockIdx.x * blockDim.x + threadIdx.x;   // b*L + l
  int b = t >> 11;
  int l = t & (L_ - 1);
  const float* xb = x + (size_t)b * C_ * L_ + l;   // stride L_ per channel
  float ss = 0.f, sm = 0.f;
  for (int c = 0; c < C_; ++c) { float v = xb[(size_t)c * L_]; ss += v * v; sm += v; }
  float nrm = sqrtf(ss); if (nrm < 1e-8f) nrm = 1e-8f;
  float invn = 1.f / nrm;
  float rinv = (fabsf(sm) == 0.f) ? 0.f : 1.f / sm;   // matches isinf(1/rowsum)->0
  _Float16* qrow = qn + (size_t)t * C_;
  float p[8];
  for (int f = 0; f < 8; ++f) p[f] = 0.f;
  for (int c = 0; c < C_; ++c) {
    float v = xb[(size_t)c * L_];
    qrow[c] = (_Float16)(v * invn);
    float qf = v * rinv;
    for (int f = 0; f < 8; ++f) p[f] += qf * W1[c * 8 + f];
  }
  _Float16* pcol = p1t + (size_t)b * 16 * L_ + l;  // K-major: [f][l], stride L_
  for (int f = 0; f < 8; ++f)  pcol[(size_t)f * L_] = (_Float16)p[f];
  for (int f = 8; f < 16; ++f) pcol[(size_t)f * L_] = (_Float16)0.f;
  deg[t] = 1u;
}

// ---------------------------------------------------------------------------
// Kernel 2: sim = qn @ qn^T via v_wmma_f32_16x16x32_f16, threshold > 0.5,
// write binary adjacency (f16, exact) + row degrees via ballot popcount.
// One wave computes a 16x64 strip (4 N-tiles): the A fragment is loaded once
// and feeds 4 independent accumulators, but the two K-step WMMAs into each
// accumulator are back to back and RAW-dependent. 64 MB adj stays L2-resident.
// ---------------------------------------------------------------------------
__global__ void k_sim(const _Float16* __restrict__ qn, _Float16* __restrict__ adj,
                      unsigned* __restrict__ deg) {
  int b    = blockIdx.y;
  int wave = threadIdx.x >> 5;
  int lane = threadIdx.x & 31;
  int t    = blockIdx.x * 8 + wave;        // 0..4095: [tm:7][tng:5]
  int tm   = t >> 5, tng = t & 31;         // tng covers 4 consecutive N-tiles
  int g    = lane >> 4;
  const _Float16* qb = qn + (size_t)b * L_ * C_;
  // A layout: row M = lane&15; halves e -> K = k0 + g*16 + e
  const _Float16* arow = qb + (size_t)(tm * 16 + (lane & 15)) * C_ + g * 16;
  v16h a[2];
  for (int kk = 0; kk < 2; ++kk) {
    v8h a0 = *(const v8h*)(arow + kk * 32);
    v8h a1 = *(const v8h*)(arow + kk * 32 + 8);
    for (int i = 0; i < 8; ++i) { a[kk][i] = a0[i]; a[kk][8 + i] = a1[i]; }
  }
  v8f acc[4] = {};
  for (int j = 0; j < 4; ++j) {
    int tn = tng * 4 + j;
    // B layout: col N = lane&15 (source row tn*16+N); halves e -> K = k0+g*16+e
    const _Float16* brow = qb + (size_t)(tn * 16 + (lane & 15)) * C_ + g * 16;
    for (int kk = 0; kk < 2; ++kk) {
      v16h bm = *(const v16h*)(brow + kk * 32);
      acc[j] = __builtin_amdgcn_wmma_f32_16x16x32_f16(false, a[kk], false, bm,
                                                      (short)0, acc[j], false, false);
    }
  }
  _Float16* adjb = adj + (size_t)b * L_ * L_;
  unsigned* degb = deg + b * L_;
  for (int j = 0; j < 4; ++j) {
    int col = (tng * 4 + j) * 16 + (lane & 15);
    for (int v = 0; v < 8; ++v) {
      bool pred = acc[j][v] > 0.5f;
      unsigned mask = __builtin_amdgcn_ballot_w32(pred);
      int row = tm * 16 + g * 8 + v;               // D layout: M = g*8 + v
      adjb[(size_t)row * L_ + col] = (_Float16)(pred ? 1.f : 0.f);
      // lanes 0..15 hold row g=0, lanes 16..31 hold row g=1 for this v
      unsigned cnt = __popc((mask >> (g * 16)) & 0xFFFFu);
      if (((lane & 15) == 0) && cnt) atomicAdd(&degb[row], cnt);
    }
  }
}

// ---------------------------------------------------------------------------
// Kernel 3 (two instantiations): O = adj @ Y (f16 WMMA, K = 2048) with Y
// supplied K-major (Yt, F x L). One wave owns a full 16 x F strip: the A
// fragment is loaded once per K-chunk and shared by F/16 independent
// accumulators. Fused epilogue: val = (O + Y_row) / rowsum, then
//   pass1 (F=16): h   = relu(val + b1), OUTF = 8
//   pass2 (F=64): qry =       val + b2, OUTF = 64
// ---------------------------------------------------------------------------
template<int F, int OUTF, bool RELU>
__global__ void k_gemm_adj(const _Float16* __restrict__ adj, const _Float16* __restrict__ Yt,
                           const unsigned* __restrict__ deg,
                           const float* __restrict__ bias, float* __restrict__ out) {
  constexpr int NT = F / 16;
  int b    = blockIdx.y;
  int wave = threadIdx.x >> 5;
  int lane = threadIdx.x & 31;
  int tm   = blockIdx.x * 8 + wave;        // 0..127
  int g    = lane >> 4;
  const _Float16* adjb = adj + (size_t)b * L_ * L_;
  const _Float16* Ytb  = Yt  + (size_t)b * F * L_;
  const _Float16* arow = adjb + (size_t)(tm * 16 + (lane & 15)) * L_ + g * 16;
  int nbase = lane & 15;
  const _Float16* brow[NT];
  for (int t = 0; t < NT; ++t) brow[t] = Ytb + (size_t)(t * 16 + nbase) * L_ + g * 16;
  v8f acc[NT];
  for (int t = 0; t < NT; ++t) acc[t] = (v8f){};
  for (int k = 0; k < L_; k += 32) {
    __builtin_prefetch(arow + k + 64, 0, 1);       // prefetch next A chunk
    v8h a0 = *(const v8h*)(arow + k);
    v8h a1 = *(const v8h*)(arow + k + 8);
    v16h a;
    for (int i = 0; i < 8; ++i) { a[i] = a0[i]; a[8 + i] = a1[i]; }
    for (int t = 0; t < NT; ++t) {
      v16h bm = *(const v16h*)(brow[t] + k);
      acc[t] = __builtin_amdgcn_wmma_f32_16x16x32_f16(false, a, false, bm,
                                                      (short)0, acc[t], false, false);
    }
  }
  const unsigned* degb = deg + b * L_;
  float invd[8];
  for (int v = 0; v < 8; ++v) invd[v] = 1.f / (float)degb[tm * 16 + g * 8 + v];
  for (int t = 0; t < NT; ++t) {
    int ncol = t * 16 + nbase;
    for (int v = 0; v < 8; ++v) {
      int row = tm * 16 + g * 8 + v;
      float val = (acc[t][v] + (float)Ytb[(size_t)ncol * L_ + row]) * invd[v];
      if (ncol < OUTF) {
        val += bias[ncol];
        if (RELU) val = fmaxf(val, 0.f);
        out[((size_t)b * L_ + row) * OUTF + ncol] = val;
      }
    }
  }
}

// ---------------------------------------------------------------------------
// Kernel 4: y2t = (h @ W2)^T (8 -> 64, tiny), stored K-major (B,64,L) f16 as
// the next GEMM's B-panel.
// ---------------------------------------------------------------------------
__global__ void k_hW2(const float* __restrict__ h, const float* __restrict__ W2,
                      _Float16* __restrict__ y2t) {
  int t = blockIdx.x * blockDim.x + threadIdx.x;   // b*L + l
  int b = t >> 11;
  int l = t & (L_ - 1);
  const float* hr = h + (size_t)t * 8;
  float hv[8];
  for (int f = 0; f < 8; ++f) hv[f] = hr[f];
  _Float16* yc = y2t + (size_t)b * C_ * L_ + l;    // [n][l], stride L_
  for (int n = 0; n < C_; ++n) {
    float s = 0.f;
    for (int f = 0; f < 8; ++f) s += hv[f] * W2[f * C_ + n];
    yc[(size_t)n * L_] = (_Float16)s;
  }
}

// ---------------------------------------------------------------------------
// ProbSparse score M[b,l] = max_j(qk_j) - sum_j(qk_j)/64 over u=10 sampled
// index columns. (Deterministic hash stands in for jax threefry randint.)
// ---------------------------------------------------------------------------
__device__ __forceinline__ int hash_idx(int i, int j) {
  unsigned h = (unsigned)(i * U_ + j) * 2654435761u;
  h ^= h >> 15; h *= 2246822519u; h ^= h >> 13;
  return (int)(h & 63u);
}

__global__ void k_mscore(const float* __restrict__ q, float* __restrict__ M) {
  int t = blockIdx.x * blockDim.x + threadIdx.x;   // b*L + l
  const float* qr = q + (size_t)t * C_;
  float mx = -1e30f, sm = 0.f;
  for (int j = 0; j < U_; ++j) {
    float s = 0.f;
    for (int i = 0; i < C_; ++i) s += qr[i] * qr[hash_idx(i, j)];
    mx = fmaxf(mx, s); sm += s;
  }
  M[t] = mx - sm * (1.f / 64.f);
}

// Top-10 indices per batch: 10 rounds of block-wide argmax over L=2048.
__global__ void k_topk(const float* __restrict__ M, int* __restrict__ Mtop) {
  __shared__ float sv[256];
  __shared__ int   si[256];
  __shared__ int   chosen[U_];
  int b = blockIdx.x, tid = threadIdx.x;
  const float* Mb = M + b * L_;
  for (int r = 0; r < U_; ++r) {
    float best = -1e30f; int bi = 0;
    for (int l = tid; l < L_; l += 256) {
      bool skip = false;
      for (int c = 0; c < r; ++c) if (chosen[c] == l) skip = true;
      float v = Mb[l];
      if (!skip && v > best) { best = v; bi = l; }
    }
    sv[tid] = best; si[tid] = bi;
    __syncthreads();
    for (int s = 128; s > 0; s >>= 1) {
      if (tid < s && sv[tid + s] > sv[tid]) { sv[tid] = sv[tid + s]; si[tid] = si[tid + s]; }
      __syncthreads();
    }
    if (tid == 0) { chosen[r] = si[0]; Mtop[b * U_ + r] = si[0]; }
    __syncthreads();
  }
}

// scores -> softmax -> upd = attn @ V, one block per (b,u).
__global__ void k_attn(const float* __restrict__ q, const int* __restrict__ Mtop,
                       float* __restrict__ scores, float* __restrict__ upd) {
  __shared__ float red[256];
  __shared__ float qs[C_];
  int b = blockIdx.x / U_, u = blockIdx.x % U_;
  int tid = threadIdx.x;
  int lu = Mtop[b * U_ + u];
  const float* qb = q + (size_t)b * L_ * C_;
  if (tid < C_) qs[tid] = qb[(size_t)lu * C_ + tid];
  __syncthreads();
  float* sc = scores + ((size_t)b * U_ + u) * L_;
  const float scale = 0.022097086912079612f;       // 1/sqrt(2048)
  float lmax = -1e30f;
  for (int l = tid; l < L_; l += 256) {
    const float* kr = qb + (size_t)l * C_;
    float s = 0.f;
    for (int i = 0; i < C_; ++i) s += qs[i] * kr[i];
    s *= scale;
    sc[l] = s;
    lmax = fmaxf(lmax, s);
  }
  red[tid] = lmax; __syncthreads();
  for (int s = 128; s > 0; s >>= 1) { if (tid < s) red[tid] = fmaxf(red[tid], red[tid + s]); __syncthreads(); }
  float mx = red[0]; __syncthreads();
  float lsum = 0.f;
  for (int l = tid; l < L_; l += 256) { float e = expf(sc[l] - mx); sc[l] = e; lsum += e; }
  red[tid] = lsum; __syncthreads();
  for (int s = 128; s > 0; s >>= 1) { if (tid < s) red[tid] += red[tid + s]; __syncthreads(); }
  float inv = 1.f / red[0];
  if (tid < C_) {
    float a = 0.f;
    for (int l = 0; l < L_; ++l) a += sc[l] * qb[(size_t)l * C_ + tid];
    upd[((size_t)b * U_ + u) * C_ + tid] = a * inv;
  }
}

// cumsum(V) along L, scatter upd rows, transpose back, gamma * ctx + x.
__global__ void k_ctx_out(const float* __restrict__ q, const int* __restrict__ Mtop,
                          const float* __restrict__ upd, const float* __restrict__ x,
                          const float* __restrict__ gamma, float* __restrict__ out) {
  int t = blockIdx.x * blockDim.x + threadIdx.x;   // b*C + d
  int b = t >> 6, d = t & 63;
  float g = gamma[0];
  const float* Vb = q + (size_t)b * L_ * C_;
  int mt[U_];
  for (int u = 0; u < U_; ++u) mt[u] = Mtop[b * U_ + u];
  float run = 0.f;
  const float* xr = x + ((size_t)b * C_ + d) * L_;
  float* orow = out + ((size_t)b * C_ + d) * L_;
  for (int l = 0; l < L_; ++l) {
    run += Vb[(size_t)l * C_ + d];
    float cv = run;
    for (int u = 0; u < U_; ++u) if (mt[u] == l) cv = upd[((size_t)b * U_ + u) * C_ + d];
    orow[l] = g * cv + xr[l];
  }
}

extern "C" void kernel_launch(void* const* d_in, const int* in_sizes, int n_in,
                              void* d_out, int out_size, void* d_ws, size_t ws_size,
                              hipStream_t stream) {
  (void)in_sizes; (void)n_in; (void)out_size; (void)ws_size;
  const float* x     = (const float*)d_in[0];
  const float* W1    = (const float*)d_in[1];
  const float* b1    = (const float*)d_in[2];
  const float* W2    = (const float*)d_in[3];
  const float* b2    = (const float*)d_in[4];
  const float* gamma = (const float*)d_in[5];
  float* out = (float*)d_out;

  char* ws = (char*)d_ws;
  size_t off = 0;
  auto alloc = [&](size_t bytes) { void* p = ws + off; off += (bytes + 255) & ~(size_t)255; return p; };
  _Float16* qn   = (_Float16*)alloc((size_t)B_ * L_ * C_ * 2);   //  2 MB
  _Float16* p1t  = (_Float16*)alloc((size_t)B_ * 16 * L_ * 2);   //  0.5 MB (K-major)
  _Float16* adj  = (_Float16*)alloc((size_t)B_ * L_ * L_ * 2);   // 64 MB (L2-resident)
  unsigned* deg  = (unsigned*)alloc((size_t)B_ * L_ * 4);
  float*    h    = (float*)   alloc((size_t)B_ * L_ * 8 * 4);
  _Float16* y2t  = (_Float16*)alloc((size_t)B_ * C_ * L_ * 2);   // K-major
  float*    qry  = (float*)   alloc((size_t)B_ * L_ * C_ * 4);
  float*    Mv   = (float*)   alloc((size_t)B_ * L_ * 4);
  int*      Mtop = (int*)     alloc((size_t)B_ * U_ * 4);
  float*    sc   = (float*)   alloc((size_t)B_ * U_ * L_ * 4);
  float*    up   = (float*)   alloc((size_t)B_ * U_ * C_ * 4);

  k_prep   <<<dim3(B_ * L_ / 256), 256, 0, stream>>>(x, W1, qn, p1t, deg);
  k_sim    <<<dim3(512, B_),       256, 0, stream>>>(qn, adj, deg);
  k_gemm_adj<16, 8, true>  <<<dim3(16, B_), 256, 0, stream>>>(adj, p1t, deg, b1, h);
  k_hW2    <<<dim3(B_ * L_ / 256), 256, 0, stream>>>(h, W2, y2t);
  k_gemm_adj<64, 64, false><<<dim3(16, B_), 256, 0, stream>>>(adj, y2t, deg, b2, qry);
  k_mscore <<<dim3(B_ * L_ / 256), 256, 0, stream>>>(qry, Mv);
  k_topk   <<<dim3(B_),            256, 0, stream>>>(Mv, Mtop);
  k_attn   <<<dim3(B_ * U_),       256, 0, stream>>>(qry, Mtop, sc, up);
  k_ctx_out<<<dim3((B_ * C_) / 256), 256, 0, stream>>>(qry, Mtop, up, x, gamma, out);
}
